// ColorCNN_36498632081954
// MI455X (gfx1250) — matrix operand order repaired and checked by reading, not compiled, not run on hardware
//
#include <hip/hip_runtime.h>
#include <stdint.h>

typedef __attribute__((ext_vector_type(16))) __bf16 v16bf;
typedef __attribute__((ext_vector_type(8)))  float  v8f;

#define HW    65536
#define NCF   64
#define NHID  512
#define NK    64
#define NB    8
#define NPIX  524288   // NB*HW

// ---------------- init: repack weights to bf16 fragment order, zero accums ----
__global__ __launch_bounds__(256) void colorcnn_init(
    const float* __restrict__ w1, const float* __restrict__ w2,
    __bf16* __restrict__ w1b, __bf16* __restrict__ w2b,
    float* __restrict__ g_msum, int* __restrict__ g_mmax,
    float* __restrict__ g_wc)
{
  int i = blockIdx.x * 256 + threadIdx.x;           // 0..32767
  if (i < 32768) {
    int j    = i & 15;
    int lane = (i >> 4) & 31;
    int nl   = lane & 15;
    int hf   = lane >> 4;
    {   // W1 (64x512): B-frag layout, index = ((nt*2+ks)*32 + lane)*16 + j
      int ks = (i >> 9) & 1;
      int nt = i >> 10;                              // 0..31
      int k  = ks * 32 + hf * 8 + j + ((j >= 8) ? 8 : 0); // 0..63
      int n  = nt * 16 + nl;                         // 0..511
      w1b[i] = (__bf16)w1[k * NHID + n];
    }
    {   // W2 (512x64): index = ((nt*16+kk)*32 + lane)*16 + j
      int kk = (i >> 9) & 15;
      int nt = i >> 13;                              // 0..3
      int k  = kk * 32 + hf * 8 + j + ((j >= 8) ? 8 : 0); // 0..511
      int n  = nt * 16 + nl;                         // 0..63
      w2b[i] = (__bf16)w2[k * NK + n];
    }
  }
  if (i < NB * NK)      { g_msum[i] = 0.f; g_mmax[i] = 0; }
  if (i < NB * 3 * NK)  { g_wc[i] = 0.f; }
}

// ---------------- main: fused MLP + softmax + reductions (WMMA bf16) ---------
// One wave owns TWO 16-pixel M-tiles: B fragments are shared between the
// tiles (half the L2 traffic) and their WMMAs are independent, filling the
// bf16 WMMA result-hazard slots instead of v_nops.
__global__ __launch_bounds__(128) void colorcnn_main(
    const float* __restrict__ img,  const float* __restrict__ feat,
    const float* __restrict__ b1,   const float* __restrict__ b2,
    const __bf16* __restrict__ w1b, const __bf16* __restrict__ w2b,
    float* __restrict__ g_msum, int* __restrict__ g_mmax,
    float* __restrict__ g_wc,   float* __restrict__ gmask)
{
  __shared__ __bf16 lds_h[4][2][16][32];  // per-wave, per-tile H chunk (transpose)
  __shared__ float  lds_img[4][3][32];    // per-wave img tile (32 pixels)
  __shared__ float  lds_msum[NK];
  __shared__ int    lds_mmax[NK];
  __shared__ float  lds_wc[3][NK];

  const int tid  = threadIdx.x;
  const int wave = tid >> 5;
  const int lane = tid & 31;
  const int nl   = lane & 15;
  const int half = lane >> 4;

  if (tid < NK)     { lds_msum[tid] = 0.f; lds_mmax[tid] = 0; }
  if (tid < 3 * NK) { (&lds_wc[0][0])[tid] = 0.f; }
  __syncthreads();

  const int  b       = blockIdx.x >> 9;                  // 512 blocks / image
  const long pixbase = (long)blockIdx.x * 128 + wave * 32; // 32 pixels / wave
  const int  pix0    = (int)(pixbase & (HW - 1));

  // img tile (3 x 32) to LDS
  #pragma unroll
  for (int c = 0; c < 3; ++c)
    lds_img[wave][c][lane] = img[((long)(b * 3 + c)) * HW + pix0 + lane];

  // A fragments of X (2 tiles x 16 pixels x 64 ch) from feat, f32->bf16
  const float* fb = feat + ((long)b * NCF) * HW + pix0;
  v16bf a0[2], a1[2];
  #pragma unroll
  for (int j = 0; j < 16; ++j) {
    int kl = j + half * 8 + ((j >= 8) ? 8 : 0);          // 16-bit A layout
    #pragma unroll
    for (int pt = 0; pt < 2; ++pt) {
      a0[pt][j] = (__bf16)fb[(long)kl * HW + pt * 16 + nl];
      a1[pt][j] = (__bf16)fb[(long)(kl + 32) * HW + pt * 16 + nl];
    }
  }

  // logits accumulators (2 tiles x 16x64), init with bias b2[n]
  v8f acc[2][4];
  #pragma unroll
  for (int t = 0; t < 4; ++t) {
    float bv = b2[t * 16 + nl];
    v8f v = {bv, bv, bv, bv, bv, bv, bv, bv};
    acc[0][t] = v;
    acc[1][t] = v;
  }

  // fused GEMM1 -> ReLU -> GEMM2, 32-wide hidden chunks
  for (int kk = 0; kk < 16; ++kk) {
    #pragma unroll
    for (int s = 0; s < 2; ++s) {
      int nt = kk * 2 + s;
      float hb = b1[nt * 16 + nl];
      v8f hA = {hb, hb, hb, hb, hb, hb, hb, hb};
      v8f hB = hA;
      v16bf bw0 = *(const v16bf*)(w1b + ((nt * 2 + 0) * 32 + lane) * 16);
      v16bf bw1 = *(const v16bf*)(w1b + ((nt * 2 + 1) * 32 + lane) * 16);
      // interleaved independent WMMAs (tile A / tile B share B operands)
      hA = __builtin_amdgcn_wmma_f32_16x16x32_bf16(false, a0[0], false, bw0,
                                                   (short)0, hA, false, false);
      hB = __builtin_amdgcn_wmma_f32_16x16x32_bf16(false, a0[1], false, bw0,
                                                   (short)0, hB, false, false);
      hA = __builtin_amdgcn_wmma_f32_16x16x32_bf16(false, a1[0], false, bw1,
                                                   (short)0, hA, false, false);
      hB = __builtin_amdgcn_wmma_f32_16x16x32_bf16(false, a1[1], false, bw1,
                                                   (short)0, hB, false, false);
      // ReLU + D-layout -> A-layout transpose through LDS (bf16)
      #pragma unroll
      for (int r = 0; r < 8; ++r) {
        lds_h[wave][0][r + 8 * half][s * 16 + nl] = (__bf16)fmaxf(hA[r], 0.f);
        lds_h[wave][1][r + 8 * half][s * 16 + nl] = (__bf16)fmaxf(hB[r], 0.f);
      }
    }
    asm volatile("s_wait_dscnt 0" ::: "memory");
    v16bf a2A, a2B;
    #pragma unroll
    for (int j = 0; j < 16; ++j) {
      int kl = j + half * 8 + ((j >= 8) ? 8 : 0);
      a2A[j] = lds_h[wave][0][nl][kl];
      a2B[j] = lds_h[wave][1][nl][kl];
    }
    #pragma unroll
    for (int t = 0; t < 4; ++t) {
      v16bf bw = *(const v16bf*)(w2b + ((t * 16 + kk) * 32 + lane) * 16);
      acc[0][t] = __builtin_amdgcn_wmma_f32_16x16x32_bf16(false, a2A, false, bw,
                                                          (short)0, acc[0][t], false, false);
      acc[1][t] = __builtin_amdgcn_wmma_f32_16x16x32_bf16(false, a2B, false, bw,
                                                          (short)0, acc[1][t], false, false);
    }
  }

  // per-tile: softmax over K=64, then store mask + reductions
  const int mrow = 8 * half;
  #pragma unroll
  for (int pt = 0; pt < 2; ++pt) {
    // softmax per pixel row (VGPR row r <-> pixel m = r + 8*half)
    #pragma unroll
    for (int r = 0; r < 8; ++r) {
      float mx = acc[pt][0][r];
      #pragma unroll
      for (int t = 1; t < 4; ++t) mx = fmaxf(mx, acc[pt][t][r]);
      #pragma unroll
      for (int off = 1; off < 16; off <<= 1)
        mx = fmaxf(mx, __shfl_xor(mx, off, 32));
      float sum = 0.f;
      #pragma unroll
      for (int t = 0; t < 4; ++t) {
        float e = __expf(acc[pt][t][r] - mx);
        acc[pt][t][r] = e;
        sum += e;
      }
      #pragma unroll
      for (int off = 1; off < 16; off <<= 1)
        sum += __shfl_xor(sum, off, 32);
      float inv = 1.f / sum;
      #pragma unroll
      for (int t = 0; t < 4; ++t) acc[pt][t][r] *= inv;
    }

    // store mask + accumulate sum / max / weighted-color
    #pragma unroll
    for (int t = 0; t < 4; ++t) {
      float ssum = 0.f, smax = 0.f;
      float wcp0 = 0.f, wcp1 = 0.f, wcp2 = 0.f;
      #pragma unroll
      for (int r = 0; r < 8; ++r) {
        float v = acc[pt][t][r];
        ssum += v;
        smax  = fmaxf(smax, v);
        wcp0 += lds_img[wave][0][pt * 16 + r + mrow] * v;
        wcp1 += lds_img[wave][1][pt * 16 + r + mrow] * v;
        wcp2 += lds_img[wave][2][pt * 16 + r + mrow] * v;
        gmask[(pixbase + pt * 16 + r + mrow) * NK + t * 16 + nl] = v;
      }
      ssum += __shfl_xor(ssum, 16, 32);
      smax  = fmaxf(smax, __shfl_xor(smax, 16, 32));
      wcp0 += __shfl_xor(wcp0, 16, 32);
      wcp1 += __shfl_xor(wcp1, 16, 32);
      wcp2 += __shfl_xor(wcp2, 16, 32);
      if (lane < 16) {
        atomicAdd(&lds_msum[t * 16 + nl], ssum);
        atomicMax(&lds_mmax[t * 16 + nl], __float_as_int(smax)); // mask > 0
        atomicAdd(&lds_wc[0][t * 16 + nl], wcp0);
        atomicAdd(&lds_wc[1][t * 16 + nl], wcp1);
        atomicAdd(&lds_wc[2][t * 16 + nl], wcp2);
      }
    }
  }
  __syncthreads();
  if (tid < NK) {
    atomicAdd(&g_msum[b * NK + tid], lds_msum[tid]);
    atomicMax(&g_mmax[b * NK + tid], lds_mmax[tid]);
    #pragma unroll
    for (int c = 0; c < 3; ++c)
      atomicAdd(&g_wc[(b * 3 + c) * NK + tid], lds_wc[c][tid]);
  }
}

// ---------------- pass 2: transformed_img = mask @ weighted_color ------------
__global__ __launch_bounds__(256) void colorcnn_transform(
    const float* __restrict__ gmask, const float* __restrict__ g_wc,
    float* __restrict__ out)
{
  __shared__ float wcl[3 * NK];
  const long p = (long)blockIdx.x * 256 + threadIdx.x;    // global pixel
  const int  b = (int)(p >> 16);
  if (threadIdx.x < 3 * NK)
    wcl[threadIdx.x] = g_wc[b * 3 * NK + threadIdx.x] * (1.0f / (float)HW);
  __syncthreads();

  const float4* mrow = (const float4*)gmask + p * 16;
  float a0 = 0.f, a1 = 0.f, a2 = 0.f;
  #pragma unroll
  for (int q = 0; q < 16; ++q) {
    float4 mv = mrow[q];
    int k = q * 4;
    a0 += mv.x * wcl[k]          + mv.y * wcl[k + 1]
        + mv.z * wcl[k + 2]      + mv.w * wcl[k + 3];
    a1 += mv.x * wcl[64 + k]     + mv.y * wcl[64 + k + 1]
        + mv.z * wcl[64 + k + 2] + mv.w * wcl[64 + k + 3];
    a2 += mv.x * wcl[128 + k]    + mv.y * wcl[128 + k + 1]
        + mv.z * wcl[128 + k + 2]+ mv.w * wcl[128 + k + 3];
  }
  const int hw = (int)(p & (HW - 1));
  out[((long)(b * 3 + 0)) * HW + hw] = a0;
  out[((long)(b * 3 + 1)) * HW + hw] = a1;
  out[((long)(b * 3 + 2)) * HW + hw] = a2;
}

// ---------------- pass 3: scalars --------------------------------------------
__global__ void colorcnn_finalize(const float* __restrict__ g_msum,
                                  const int* __restrict__ g_mmax,
                                  float* __restrict__ out_scalars)
{
  if (threadIdx.x != 0 || blockIdx.x != 0) return;
  float mm = 0.f;
  for (int i = 0; i < NB * NK; ++i) mm += __int_as_float(g_mmax[i]);
  mm *= (1.f / (float)(NB * NK));
  float sm = 0.f;
  for (int bb = 0; bb < NB; ++bb) {
    float mu[NK];
    float mean = 0.f;
    for (int k = 0; k < NK; ++k) {
      mu[k] = g_msum[bb * NK + k] * (1.f / (float)HW);
      mean += mu[k];
    }
    mean *= (1.f / (float)NK);
    float var = 0.f;
    for (int k = 0; k < NK; ++k) { float d = mu[k] - mean; var += d * d; }
    var *= (1.f / (float)(NK - 1));        // ddof=1
    sm += sqrtf(var);
  }
  sm *= (1.f / (float)NB);
  out_scalars[0] = mm;
  out_scalars[1] = sm;
}

// ---------------- launcher ---------------------------------------------------
extern "C" void kernel_launch(void* const* d_in, const int* in_sizes, int n_in,
                              void* d_out, int out_size, void* d_ws, size_t ws_size,
                              hipStream_t stream) {
  const float* img  = (const float*)d_in[0];   // (8,3,256,256)
  const float* feat = (const float*)d_in[1];   // (8,64,256,256)
  // d_in[2] coord_map unused
  const float* w1   = (const float*)d_in[3];   // (64,512)
  const float* b1   = (const float*)d_in[4];   // (512)
  const float* w2   = (const float*)d_in[5];   // (512,64)
  const float* b2   = (const float*)d_in[6];   // (64)
  float* out = (float*)d_out;

  uint8_t* ws = (uint8_t*)d_ws;
  __bf16* w1b   = (__bf16*)(ws);                 //  64 KB
  __bf16* w2b   = (__bf16*)(ws + (64 << 10));    //  64 KB
  float*  g_msum = (float*)(ws + (128 << 10));   //   2 KB
  int*    g_mmax = (int*)  (ws + (130 << 10));   //   2 KB
  float*  g_wc   = (float*)(ws + (132 << 10));   //   6 KB
  float*  gmask  = (float*)(ws + (144 << 10));   // 128 MB (NPIX * 64 f32)

  colorcnn_init<<<128, 256, 0, stream>>>(w1, w2, w1b, w2b, g_msum, g_mmax, g_wc);
  colorcnn_main<<<NPIX / 128, 128, 0, stream>>>(img, feat, b1, b2, w1b, w2b,
                                                g_msum, g_mmax, g_wc, gmask);
  colorcnn_transform<<<NPIX / 256, 256, 0, stream>>>(gmask, g_wc, out);
  colorcnn_finalize<<<1, 32, 0, stream>>>(g_msum, g_mmax, out + (long)NB * 3 * HW);
}
